// QuantLinearMLX_90761248899767
// MI455X (gfx1250) — matrix operand order repaired and checked by reading, not compiled, not run on hardware
//
#include <hip/hip_runtime.h>

// ---- CDNA5 vector types ----
typedef __attribute__((ext_vector_type(16))) _Float16 v16h;
typedef __attribute__((ext_vector_type(8)))  _Float16 v8h;
typedef __attribute__((ext_vector_type(4)))  _Float16 v4h;
typedef __attribute__((ext_vector_type(8)))  float    v8f;
typedef __attribute__((ext_vector_type(4)))  float    v4f;
typedef __attribute__((ext_vector_type(4)))  unsigned int v4u;

#define KDIM   4096
#define NDIM   11008
#define MDIM   8192          // 4 * 2048 tokens
#define KWORDS (KDIM / 8)    // 512 packed u32 per output row
#define GROUPS (KDIM / 128)  // 32 quant groups per row

#define BM 256
#define BN 128
#define BK 32
#define PAD 8
#define LDSS (BK + PAD)      // 40 halves = 80 B row stride: 16B-aligned + bank-conflict-free
#define STAGES (KDIM / BK)   // 128 (even)
#define NTHREADS 512         // 16 wave32s: 4 (M) x 4 (N)

#define SA_BUF (BM * LDSS)   // elements per A buffer
#define SB_BUF (BN * LDSS)   // elements per B buffer

__global__ __launch_bounds__(NTHREADS)
void qlinear_wmma_kernel(const float* __restrict__ x,
                         const unsigned int* __restrict__ qweight,
                         const _Float16* __restrict__ scales,
                         const _Float16* __restrict__ biases,
                         const _Float16* __restrict__ lin_bias,
                         float* __restrict__ out)
{
    __shared__ _Float16 sA[2 * SA_BUF];   // activations f16, [M][K] row-major, double buffered
    __shared__ _Float16 sB[2 * SB_BUF];   // dequant weights f16, [N][K] row-major, double buffered

    const int tid  = threadIdx.x;
    const int r    = (tid & 31) & 15;    // fragment row / N column
    const int lh   = (tid & 31) >> 4;    // lane half -> K sub-range
    const int wave = tid >> 5;
    const int wm   = wave & 3;           // 4 waves along M (64 rows each)
    const int wn   = wave >> 2;          // 4 waves along N (32 cols each)

    const int m0 = blockIdx.y * BM;
    const int n0 = blockIdx.x * BN;

    // ---- accumulators: 4(M) x 2(N) subtiles of 16x16 per wave ----
    v8f acc[8];
#pragma unroll
    for (int i = 0; i < 8; ++i) acc[i] = (v8f){};

    // ---- per-thread hoisted base pointers ----
    // X: 2048 float4 positions (256 rows x 8); pos = tid + q*512 -> same col for all q
    const int xrow = tid >> 3;                 // + 64*q
    const int xcol = (tid & 7) * 4;
    const float* xbase = x + (size_t)(m0 + xrow) * KDIM + xcol;

    // W: 512 words (128 rows x 4 words); 1 word per thread
    const int nl = tid >> 2;                   // 0..127
    const int kw = tid & 3;                    // word-in-stage
    const unsigned int* wbase = qweight + (size_t)(n0 + nl) * KWORDS + kw;
    const _Float16* scrow = scales + (size_t)(n0 + nl) * GROUPS;
    const _Float16* birow = biases + (size_t)(n0 + nl) * GROUPS;

    _Float16* sBst = &sB[nl * LDSS + kw * 8];
    _Float16* sAst = &sA[xrow * LDSS + xcol];

    // fragment read pointers (buffer select added as compile-time constant)
    const _Float16* apA = &sA[(wm * 64 + r) * LDSS + lh * 8];    // +i*16*LDSS
    const _Float16* apB = &sB[(wn * 32 + r) * LDSS + lh * 8];    // +j*16*LDSS

    // ---- stage registers ----
    v4f xr[4];
    unsigned int wword;
    _Float16 wsc, wbi;

    auto load_stage = [&](int s) {
        const int k0 = s * BK;
#pragma unroll
        for (int q = 0; q < 4; ++q)
            xr[q] = *(const v4f*)(xbase + (size_t)q * 64 * KDIM + k0);
        wword = wbase[s * 4];
        const int g = s >> 2;                  // group = k0/128
        wsc = scrow[g];
        wbi = birow[g];
    };

    auto store_stage = [&](int buf) {
        // X: f32 -> f16
#pragma unroll
        for (int q = 0; q < 4; ++q)
            *(v4h*)(sAst + buf * SA_BUF + q * 64 * LDSS) = __builtin_convertvector(xr[q], v4h);

        // W: 8 nibbles -> 8 f16 via (q | 0x6400) = 1024+q trick, packed math.
        const unsigned int w  = wword;
        const unsigned int ev = w & 0x0F0F0F0Fu;          // nibbles 0,2,4,6
        const unsigned int od = (w >> 4) & 0x0F0F0F0Fu;   // nibbles 1,3,5,7
        union { unsigned int u[4]; v8h h; } t;
#pragma unroll
        for (int p = 0; p < 4; ++p) {
            // byte0 = ev.byte(p) (sel p from src1), byte2 = od.byte(p) (sel 4+p from src0)
            unsigned int sel = 0x0C040C00u | (unsigned)(p * 0x00010001);
            t.u[p] = __builtin_amdgcn_perm(od, ev, sel) | 0x64006400u;
        }
        v8h qv = t.h + (_Float16)(-1024.0f);              // exact: (1024+q)-1024 = q
        v8h wv = qv * wsc + wbi;                          // v_pk_fma_f16 (splat scale/bias)
        *(v8h*)(sBst + buf * SB_BUF) = wv;                // ds_store_b128
    };

    auto compute = [&](int buf) {
        // A fragments: 16-bit A layout — lane<16: K{0..7,16..23}; lane>=16: K{8..15,24..31}
        v16h af[4];
#pragma unroll
        for (int i = 0; i < 4; ++i) {
            const _Float16* p = apA + buf * SA_BUF + i * 16 * LDSS;
            union { v4u u[2]; v16h v; } t;
            t.u[0] = *(const v4u*)p;            // ds_load_b128
            t.u[1] = *(const v4u*)(p + 16);     // ds_load_b128
            af[i] = t.v;
        }
        // B fragments: lane = N column; lane<16 holds K{0..7,16..23}, lane>=16 holds K{8..15,24..31}
        v16h bf[2];
#pragma unroll
        for (int j = 0; j < 2; ++j) {
            const _Float16* p = apB + buf * SB_BUF + j * 16 * LDSS;
            union { v4u u[2]; v16h v; } t;
            t.u[0] = *(const v4u*)p;
            t.u[1] = *(const v4u*)(p + 16);
            bf[j] = t.v;
        }
#pragma unroll
        for (int i = 0; i < 4; ++i)
#pragma unroll
            for (int j = 0; j < 2; ++j)
                acc[i * 2 + j] = __builtin_amdgcn_wmma_f32_16x16x32_f16(
                    false, af[i], false, bf[j], (short)0, acc[i * 2 + j], false, false);
    };

    // ---------------- prologue ----------------
    load_stage(0);
    store_stage(0);          // buf0 <- stage 0
    load_stage(1);           // regs <- stage 1 (in flight)
    __syncthreads();

    // ---------------- main K loop: 2 stages per trip, 1 barrier per stage ----------------
    for (int s = 0; s < STAGES; s += 2) {
        // stage s from buf0; regs hold stage s+1
        store_stage(1);                          // buf1 <- stage s+1 (loads landed last iter)
        if (s + 2 < STAGES) load_stage(s + 2);   // regs <- stage s+2 (in flight)
        if (s + 2 < STAGES) {
            __builtin_prefetch(xbase + (s + 3) * BK, 0, 0);
            __builtin_prefetch(wbase + (s + 3) * 4, 0, 0);
        }
        compute(0);
        __syncthreads();

        // stage s+1 from buf1; regs hold stage s+2
        if (s + 2 < STAGES) store_stage(0);      // buf0 <- stage s+2
        if (s + 3 < STAGES) load_stage(s + 3);   // regs <- stage s+3
        compute(1);
        __syncthreads();
    }

    // ---- epilogue: C layout — VGPR t: lane<16 -> M=t, lane>=16 -> M=8+t ----
#pragma unroll
    for (int j = 0; j < 2; ++j) {
        const int c  = n0 + wn * 32 + j * 16 + r;
        const float lb = (float)lin_bias[c];
#pragma unroll
        for (int i = 0; i < 4; ++i) {
            const int rowbase = m0 + wm * 64 + i * 16 + lh * 8;
            float* po = out + (size_t)rowbase * NDIM + c;
            v8f a = acc[i * 2 + j];
#pragma unroll
            for (int t = 0; t < 8; ++t)
                po[(size_t)t * NDIM] = a[t] + lb;
        }
    }
}

extern "C" void kernel_launch(void* const* d_in, const int* in_sizes, int n_in,
                              void* d_out, int out_size, void* d_ws, size_t ws_size,
                              hipStream_t stream) {
    (void)in_sizes; (void)n_in; (void)out_size; (void)d_ws; (void)ws_size;
    const float*        xp = (const float*)d_in[0];
    const unsigned int* qw = (const unsigned int*)d_in[1];
    const _Float16*     sc = (const _Float16*)d_in[2];
    const _Float16*     bi = (const _Float16*)d_in[3];
    const _Float16*     lb = (const _Float16*)d_in[4];
    float*              op = (float*)d_out;

    dim3 grid(NDIM / BN, MDIM / BM);   // 86 x 32 blocks
    qlinear_wmma_kernel<<<grid, NTHREADS, 0, stream>>>(xp, qw, sc, bi, lb, op);
}
